// DRNLayer_67181878444484
// MI455X (gfx1250) — matrix operand order repaired and checked by reading, not compiled, not run on hardware
//
#include <hip/hip_runtime.h>
#include <hip/hip_bf16.h>

typedef __attribute__((ext_vector_type(16))) _Float16 v16h;
typedef __attribute__((ext_vector_type(8)))  float    v8f;
typedef __attribute__((ext_vector_type(4)))  float    v4f;

#define QL     100
#define QU     100
#define NL     64
#define NU     64
#define BATCH  64

#define KPAD   128   // padded qL (K dimension), zeros beyond 100
#define TROW   136   // f16 row stride (272B = 17*16B: rows stay 16B aligned, 4-bank skew)
#define NPAD   112   // 7 tiles of 16 covering qU=100
#define MROWS  32    // batch rows per workgroup

#define LOG2E  1.4426950408889634f
#define LN2    0.6931471805599453f

// Pack two 16-byte LDS chunks into a v16h fragment (two ds_load_b128).
__device__ __forceinline__ v16h pack16(const _Float16* p0, const _Float16* p1) {
    union { v4f f[2]; v16h h; } u;
    u.f[0] = *(const v4f*)p0;
    u.f[1] = *(const v4f*)p1;
    return u.h;
}

// Raw HW transcendentals (operands are range-safe by construction).
__device__ __forceinline__ float fast_ln(float x)   { return __builtin_amdgcn_logf(x) * LN2; }
__device__ __forceinline__ float fast_exp2(float x) { return __builtin_amdgcn_exp2f(x); }

// One 16x16 N-tile: 4 B fragments from LDS, 4 chained WMMAs, log-accumulate.
__device__ __forceinline__ void tile_accum(const _Float16* __restrict__ sT,
                                           int lrow, int half,
                                           const v16h afr[4], int nt, v8f& ls) {
    const _Float16* tb = sT + (nt * 16 + lrow) * TROW;
    v16h b[4];
    #pragma unroll
    for (int ks = 0; ks < 4; ++ks) {
        int mb = ks * 32 + half * 8;             // K halves 8h..8h+7 and 16+8h..16+8h+7 per lane
        b[ks] = pack16(tb + mb, tb + mb + 16);
    }
    v8f acc = {};
    #pragma unroll
    for (int ks = 0; ks < 4; ++ks)
        acc = __builtin_amdgcn_wmma_f32_16x16x32_f16(
                  false, afr[ks], false, b[ks], (short)0, acc, false, false);
    #pragma unroll
    for (int z = 0; z < 8; ++z) {
        float v = fminf(fmaxf(acc[z], 1e-15f), 1e15f);   // v_med3
        ls[z] += fast_ln(v);
    }
}

__global__ __launch_bounds__(256)
void drn_softmax_kernel(const float* __restrict__ P,
                        const float* __restrict__ W,
                        const float* __restrict__ bias_abs,
                        const float* __restrict__ bias_q,
                        const float* __restrict__ lambda_abs,
                        const float* __restrict__ lambda_q,
                        float* __restrict__ out)
{
    __shared__ __align__(16) _Float16 sT[NPAD * TROW];    // 30464 B  T_jk tile
    __shared__ __align__(16) _Float16 sA[MROWS * TROW];   //  8704 B  P_k tile (f16)
    __shared__ __align__(16) float    sL[MROWS * NPAD];   // 14336 B  epilogue logsum

    const int j     = blockIdx.x;          // upper node
    const int ibase = blockIdx.y * MROWS;  // batch base row
    const int tid   = threadIdx.x;
    const int lane  = tid & 31;
    const int half  = (lane >> 4) & 1;
    const int lrow  = lane & 15;

    // Wave-derived indices made scalar: SALU branches keep EXEC all-1s for WMMA.
    const int swave = __builtin_amdgcn_readfirstlane(tid >> 5);  // 0..7
    const int mtl   = swave & 1;           // which 16-row batch sub-tile
    const int ntg   = swave >> 1;          // 0..3
    const bool twoTiles = (ntg < 3);       // scalar condition

    v8f ls0 = {}; v8f ls1 = {};

    for (int k = 0; k < NL; ++k) {
        __syncthreads();   // previous tiles fully consumed before overwrite

        // --- stage A: P[ibase..+31, k, :] -> f16 (zero pad m>=100) ---
        {
            const float* Pk = P + (size_t)ibase * NL * QL + (size_t)k * QL;
            #pragma unroll 4
            for (int e = tid; e < MROWS * KPAD; e += 256) {
                int r = e >> 7;
                int m = e & (KPAD - 1);
                float v = (m < QL) ? Pk[(size_t)r * NL * QL + m] : 0.0f;
                sA[r * TROW + m] = (_Float16)v;
            }
            // prefetch next k's P rows into cache (global_prefetch_b8)
            if (k + 1 < NL && tid < MROWS)
                __builtin_prefetch(Pk + (size_t)tid * NL * QL + QL, 0, 0);
        }
        // --- stage T_jk: T[l,m] = exp2(-w*log2e * ((m-l)/100)^2), zero pad ---
        {
            const float w2 = W[j * NL + k] * LOG2E;
            #pragma unroll 4
            for (int e = tid; e < NPAD * KPAD; e += 256) {
                int l = e >> 7;
                int m = e & (KPAD - 1);
                float t = 0.0f;
                if (l < QU && m < QL) {
                    float d = (float)(m - l) * 0.01f;
                    t = fast_exp2(-w2 * d * d);
                }
                sT[l * TROW + m] = (_Float16)t;
            }
        }
        __syncthreads();

        // --- A fragments for this wave's batch sub-tile (ISA 16-bit A layout) ---
        v16h afr[4];
        {
            const _Float16* abase = &sA[(mtl * 16 + lrow) * TROW];
            #pragma unroll
            for (int ks = 0; ks < 4; ++ks) {
                int mb = ks * 32 + half * 8;
                afr[ks] = pack16(abase + mb, abase + mb + 16);
            }
        }

        // --- owned N-tiles: static accumulators, scalar-guarded second tile ---
        tile_accum(sT, lrow, half, afr, ntg, ls0);
        if (twoTiles)
            tile_accum(sT, lrow, half, afr, ntg + 4, ls1);
    }

    __syncthreads();
    // --- spill logsum accumulators to LDS in C-matrix layout ---
    {
        int col = ntg * 16 + lrow;
        if (col < QU) {
            #pragma unroll
            for (int g = 0; g < 8; ++g)
                sL[(mtl * 16 + g + 8 * half) * NPAD + col] = ls0[g];
        }
        if (twoTiles) {
            col = (ntg + 4) * 16 + lrow;
            if (col < QU) {
                #pragma unroll
                for (int g = 0; g < 8; ++g)
                    sL[(mtl * 16 + g + 8 * half) * NPAD + col] = ls1[g];
            }
        }
    }
    __syncthreads();

    // --- per-row bias + stable softmax over l (qU) ---
    if (tid < MROWS) {
        const int row = tid;
        const float bq = bias_q[j],    ba = bias_abs[j];
        const float lq = lambda_q[j],  la = lambda_abs[j];
        float* rowp = &sL[row * NPAD];

        float mx = -3.0e38f;
        for (int l = 0; l < QU; ++l) {
            float s0 = (float)l * 0.01f;
            float dq = s0 - lq;
            float v  = rowp[l] - bq * dq * dq - ba * fabsf(s0 - la);
            rowp[l] = v;
            mx = fmaxf(mx, v);
        }
        float sum = 0.0f;
        for (int l = 0; l < QU; ++l) sum += fast_exp2((rowp[l] - mx) * LOG2E);
        const float inv = 1.0f / sum;
        float* orow = out + ((size_t)(ibase + row) * NU + j) * QU;
        for (int l = 0; l < QU; ++l)
            orow[l] = fast_exp2((rowp[l] - mx) * LOG2E) * inv;
    }
}

extern "C" void kernel_launch(void* const* d_in, const int* in_sizes, int n_in,
                              void* d_out, int out_size, void* d_ws, size_t ws_size,
                              hipStream_t stream) {
    const float* P          = (const float*)d_in[0];
    const float* weight     = (const float*)d_in[1];
    const float* bias_abs   = (const float*)d_in[2];
    const float* bias_q     = (const float*)d_in[3];
    const float* lambda_abs = (const float*)d_in[4];
    const float* lambda_q   = (const float*)d_in[5];
    float* out = (float*)d_out;

    dim3 grid(NU, BATCH / MROWS);   // 64 x 2 = 128 workgroups
    drn_softmax_kernel<<<grid, 256, 0, stream>>>(P, weight, bias_abs, bias_q,
                                                 lambda_abs, lambda_q, out);
}
